// ImprovedFraudGNN_6614249635872
// MI455X (gfx1250) — compile-verified, matrix operand order checked by reading, not run on hardware
//
#include <hip/hip_runtime.h>
#include <hip/hip_bf16.h>

typedef __attribute__((ext_vector_type(16))) _Float16 v16h;
typedef __attribute__((ext_vector_type(8)))  float    v8f;

// ---------------------------------------------------------------------------
// Async global->LDS staging (CDNA5 GLOBAL_LOAD_ASYNC_TO_LDS_B128, ASYNCcnt).
// Builtin takes typed vector pointers: (v4i AS1*, v4i AS3*, imm offset, cpol).
// Falls back to plain LDS stores if the builtin is unavailable.
// ---------------------------------------------------------------------------
#if __has_builtin(__builtin_amdgcn_global_load_async_to_lds_b128) && \
    __has_builtin(__builtin_amdgcn_s_wait_asynccnt)
#define GCN_USE_ASYNC_LDS 1
#else
#define GCN_USE_ASYNC_LDS 0
#endif

typedef int v4i_gnu __attribute__((vector_size(16)));
typedef __attribute__((address_space(1))) v4i_gnu* g_v4p;   // global int4*
typedef __attribute__((address_space(3))) v4i_gnu* l_v4p;   // LDS int4*

__device__ __forceinline__ g_v4p to_glb4(const void* p) {
    return (g_v4p)(unsigned long long)p;
}
__device__ __forceinline__ l_v4p to_lds4(void* p) {
    // generic LDS pointer low 32 bits == LDS offset (aperture mapping)
    return (l_v4p)(unsigned)(unsigned long long)p;
}

__device__ __forceinline__ void copy16_async(const float* g, float* l) {
#if GCN_USE_ASYNC_LDS
    __builtin_amdgcn_global_load_async_to_lds_b128(to_glb4(g), to_lds4(l), 0, 0);
#else
    *(float4*)l = *(const float4*)g;
#endif
}
__device__ __forceinline__ void wait_async_lds() {
#if GCN_USE_ASYNC_LDS
    __builtin_amdgcn_s_wait_asynccnt(0);
#endif
}

// ---------------------------------------------------------------------------
// Degree kernels: deg starts at 1.0 (self loop), +1 per incoming edge, rsqrt.
// ---------------------------------------------------------------------------
__global__ void deg_init(float* __restrict__ deg, int n) {
    int i = blockIdx.x * blockDim.x + threadIdx.x;
    if (i < n) deg[i] = 1.0f;
}

__global__ void deg_count(const int* __restrict__ dst, float* __restrict__ deg, int e) {
    int i = blockIdx.x * blockDim.x + threadIdx.x;
    if (i < e) atomicAdd(&deg[dst[i]], 1.0f);
}

__global__ void deg_finish(float* __restrict__ deg, int n) {
    int i = blockIdx.x * blockDim.x + threadIdx.x;
    if (i < n) deg[i] = rsqrtf(deg[i]);   // deg >= 1 always
}

// ---------------------------------------------------------------------------
// WMMA GEMM: H[n,FOUT] = X[n,FIN] @ W[FIN,FOUT]  (f16 inputs, f32 accumulate)
// grid.x = n/16 row tiles; blockDim.x = 32*(FOUT/16); wave w owns col tile w.
// A tile (16 x FIN) and full W staged in LDS via async global->LDS copies.
// A rows padded by 4 floats so half-wave LDS reads spread across 16 banks.
//
// VGPR layouts per CDNA5 ISA (wave32):
//   A 16x32 f16 : lane&15 = row M; lane<16 -> K in {0..7,16..23},
//                                  lane>=16 -> K in {8..15,24..31}
//   B 32x16 f16 : same K mapping, lane&15 = column N (B read column-major)
//   C/D 16x16 f32: VGPR v -> M = v + (lane<16 ? 0 : 8), N = lane&15
// ---------------------------------------------------------------------------
template <int FIN, int FOUT>
__global__ void gemm_wmma(const float* __restrict__ X, const float* __restrict__ W,
                          float* __restrict__ H) {
    constexpr int FINP = FIN + 4;              // padded LDS row stride (floats)
    __shared__ float sA[16 * FINP];
    __shared__ float sW[FIN * FOUT];

    const int tid  = threadIdx.x;
    const int nth  = 32 * (FOUT / 16);
    const int row0 = blockIdx.x * 16;

    // stage A tile: 16 rows x (FIN/4) b128 chunks  (trip counts divide nth exactly)
    constexpr int CPR = FIN / 4;
    for (int i = tid; i < 16 * CPR; i += nth) {
        const int rr = i / CPR, cc = (i % CPR) * 4;
        copy16_async(X + (size_t)(row0 + rr) * FIN + cc, sA + rr * FINP + cc);
    }
    // stage W: FIN*FOUT floats contiguous
    for (int i = tid; i < (FIN * FOUT) / 4; i += nth) {
        copy16_async(W + i * 4, sW + i * 4);
    }
    wait_async_lds();
    __syncthreads();

    const int wave  = tid >> 5;
    const int lane  = tid & 31;
    const int col0  = wave * 16;
    const int r     = lane & 15;
    const int kbase = (lane < 16) ? 0 : 8;

    v8f acc = {};
    for (int k0 = 0; k0 < FIN; k0 += 32) {
        v16h a, b;
#pragma unroll
        for (int h = 0; h < 16; ++h) {
            const int kk = kbase + h + ((h & 8) ? 8 : 0);        // ISA K mapping
            a[h] = (_Float16)sA[r * FINP + k0 + kk];
            b[h] = (_Float16)sW[(k0 + kk) * FOUT + col0 + r];
        }
        acc = __builtin_amdgcn_wmma_f32_16x16x32_f16(
                  /*neg_a=*/false, a, /*neg_b=*/false, b,
                  /*c_mod=*/(short)0, acc, /*reuse_a=*/false, /*reuse_b=*/false);
    }

    const int mbase = (lane < 16) ? 0 : 8;
#pragma unroll
    for (int v = 0; v < 8; ++v)
        H[(size_t)(row0 + mbase + v) * FOUT + (col0 + r)] = acc[v];
}

// ---------------------------------------------------------------------------
// OUT[i,:] = H[i,:] * dis[i]^2   (self-loop contribution; also initializes OUT)
// ---------------------------------------------------------------------------
template <int F>
__global__ void self_loop_init(const float* __restrict__ H, const float* __restrict__ dis,
                               float* __restrict__ OUT, long long total) {
    long long t = (long long)blockIdx.x * blockDim.x + threadIdx.x;
    if (t >= total) return;
    int i = (int)(t / F);
    float w = dis[i] * dis[i];
    OUT[t] = H[t] * w;
}

// ---------------------------------------------------------------------------
// Edge scatter: OUT[dst,:] += H[src,:] * dis[src]*dis[dst]
// F/4 lanes per edge, float4 gather (H is L2-resident), 4 f32 atomics each.
// ---------------------------------------------------------------------------
template <int F>
__global__ void scatter_edges(const float* __restrict__ H, const int* __restrict__ src,
                              const int* __restrict__ dst, const float* __restrict__ dis,
                              float* __restrict__ OUT, long long nedge) {
    constexpr int LPE = F / 4;
    long long t = (long long)blockIdx.x * blockDim.x + threadIdx.x;
    long long e = t / LPE;
    if (e >= nedge) return;
    const int fi = (int)(t % LPE) * 4;
    const int s  = src[e];
    const int d  = dst[e];
    const float norm = dis[s] * dis[d];
    const float4 hv = *(const float4*)(H + (size_t)s * F + fi);
    float* o = OUT + (size_t)d * F + fi;
    atomicAdd(o + 0, hv.x * norm);
    atomicAdd(o + 1, hv.y * norm);
    atomicAdd(o + 2, hv.z * norm);
    atomicAdd(o + 3, hv.w * norm);
}

template <int F>
__global__ void bias_relu(float* __restrict__ OUT, const float* __restrict__ b,
                          long long total) {
    long long t = (long long)blockIdx.x * blockDim.x + threadIdx.x;
    if (t >= total) return;
    float v = OUT[t] + b[(int)(t % F)];
    OUT[t] = v > 0.0f ? v : 0.0f;
}

// ---------------------------------------------------------------------------
// Classifier: out[i,c] = A2[i,:] @ Wc[:,c] + bc[c]   (32 -> 2)
// ---------------------------------------------------------------------------
__global__ void classifier(const float* __restrict__ H, const float* __restrict__ Wc,
                           const float* __restrict__ bc, float* __restrict__ out, int n) {
    int i = blockIdx.x * blockDim.x + threadIdx.x;
    if (i >= n) return;
    float a0 = bc[0], a1 = bc[1];
    const float* h = H + (size_t)i * 32;
#pragma unroll
    for (int f = 0; f < 32; ++f) {
        a0 = fmaf(h[f], Wc[f * 2 + 0], a0);
        a1 = fmaf(h[f], Wc[f * 2 + 1], a1);
    }
    out[(size_t)i * 2 + 0] = a0;
    out[(size_t)i * 2 + 1] = a1;
}

static inline long long cdivll(long long a, long long b) { return (a + b - 1) / b; }

extern "C" void kernel_launch(void* const* d_in, const int* in_sizes, int n_in,
                              void* d_out, int out_size, void* d_ws, size_t ws_size,
                              hipStream_t stream) {
    const float* x  = (const float*)d_in[0];   // [N,128]
    const int*   ei = (const int*)d_in[1];     // [2,E]
    const float* W1 = (const float*)d_in[2];   // [128,64]
    const float* b1 = (const float*)d_in[3];   // [64]
    const float* W2 = (const float*)d_in[4];   // [64,32]
    const float* b2 = (const float*)d_in[5];   // [32]
    const float* Wc = (const float*)d_in[6];   // [32,2]
    const float* bc = (const float*)d_in[7];   // [2]
    float* out = (float*)d_out;

    const int       n = in_sizes[0] / 128;     // 100000
    const long long E = in_sizes[1] / 2;       // 3200000
    const int* src = ei;
    const int* dst = ei + E;

    // Workspace carve-up (~77 MB)
    float* deg = (float*)d_ws;                 // [n]
    float* H1  = deg + n;                      // [n,64]
    float* A1  = H1 + (size_t)n * 64;          // [n,64]
    float* H2  = A1 + (size_t)n * 64;          // [n,32]
    float* A2  = H2 + (size_t)n * 32;          // [n,32]

    const int B = 256;

    // --- degrees ---
    deg_init<<<(n + B - 1) / B, B, 0, stream>>>(deg, n);
    deg_count<<<(int)cdivll(E, B), B, 0, stream>>>(dst, deg, (int)E);
    deg_finish<<<(n + B - 1) / B, B, 0, stream>>>(deg, n);

    // --- layer 1: H1 = X @ W1 (WMMA, LDS-staged) ---
    gemm_wmma<128, 64><<<n / 16, 32 * (64 / 16), 0, stream>>>(x, W1, H1);
    self_loop_init<64><<<(int)cdivll((long long)n * 64, B), B, 0, stream>>>(H1, deg, A1, (long long)n * 64);
    scatter_edges<64><<<(int)cdivll(E * 16, B), B, 0, stream>>>(H1, src, dst, deg, A1, E);
    bias_relu<64><<<(int)cdivll((long long)n * 64, B), B, 0, stream>>>(A1, b1, (long long)n * 64);

    // --- layer 2: H2 = A1 @ W2 (WMMA, LDS-staged) ---
    gemm_wmma<64, 32><<<n / 16, 32 * (32 / 16), 0, stream>>>(A1, W2, H2);
    self_loop_init<32><<<(int)cdivll((long long)n * 32, B), B, 0, stream>>>(H2, deg, A2, (long long)n * 32);
    scatter_edges<32><<<(int)cdivll(E * 8, B), B, 0, stream>>>(H2, src, dst, deg, A2, E);
    bias_relu<32><<<(int)cdivll((long long)n * 32, B), B, 0, stream>>>(A2, b2, (long long)n * 32);

    // --- classifier ---
    classifier<<<(n + B - 1) / B, B, 0, stream>>>(A2, Wc, bc, out, n);
}
